// HierarchicalMixtureOfExperts_32684701123012
// MI455X (gfx1250) — matrix operand order repaired and checked by reading, not compiled, not run on hardware
//
#include <hip/hip_runtime.h>
#include <hip/hip_bf16.h>

// ---------------------------------------------------------------------------
// Hierarchical MoE, MI455X (gfx1250). bf16 WMMA (v_wmma_f32_16x16x32_bf16)
// with fp32 accumulate; routed-sparse expert execution (top-2 of 12).
// Weights pre-converted to bf16 in d_ws when it is large enough, removing all
// v_cvt from the GEMM inner loops and halving weight traffic.
// ---------------------------------------------------------------------------

static constexpr int   N_ = 16384;
static constexpr int   D_ = 512;
static constexpr int   H_ = 1024;
static constexpr int   O_ = 512;
static constexpr int   G_ = 3;
static constexpr int   E_ = 4;
static constexpr float ALPHA_ = 0.01f;
static constexpr int   P_ = G_ * E_;   // 12 expert pairs

typedef __attribute__((ext_vector_type(16))) __bf16 v16bf;
typedef __attribute__((ext_vector_type(4)))  __bf16 v4bf;
typedef __attribute__((ext_vector_type(8)))  float  v8f;

__device__ __forceinline__ v8f wmma_bf16(v16bf a, v16bf b, v8f c) {
  // (neg_a, A, neg_b, B, c_mod, C, reuse_a, reuse_b)
  return __builtin_amdgcn_wmma_f32_16x16x32_bf16(false, a, false, b, (short)0, c,
                                                 false, false);
}

// B fragment (32x16 bf16): lane n (0..15) holds column n, K = k0..k0+15;
// lane n+16 holds column n, K = k0+16..k0+31. x @ W^T => B column n == W row n.

// fp32 weight source (fallback path): 64B -> 4x b128 loads + cvt
__device__ __forceinline__ v16bf bfrag(const float* __restrict__ p) {
  const float4* q = (const float4*)p;
  float4 f0 = q[0], f1 = q[1], f2 = q[2], f3 = q[3];
  v16bf b;
  b[0]  = (__bf16)f0.x; b[1]  = (__bf16)f0.y; b[2]  = (__bf16)f0.z; b[3]  = (__bf16)f0.w;
  b[4]  = (__bf16)f1.x; b[5]  = (__bf16)f1.y; b[6]  = (__bf16)f1.z; b[7]  = (__bf16)f1.w;
  b[8]  = (__bf16)f2.x; b[9]  = (__bf16)f2.y; b[10] = (__bf16)f2.z; b[11] = (__bf16)f2.w;
  b[12] = (__bf16)f3.x; b[13] = (__bf16)f3.y; b[14] = (__bf16)f3.z; b[15] = (__bf16)f3.w;
  return b;
}

// bf16 weight source (fast path): 32B -> 2x b128 loads, no cvt
__device__ __forceinline__ v16bf bfrag(const __bf16* __restrict__ p) {
  v16bf b;
#pragma unroll
  for (int t = 0; t < 16; ++t) b[t] = p[t];
  return b;
}

// A fragment (16x32 bf16) from row-major LDS tile, matching bfrag K order:
// lanes 0-15 (M=lane):  elems 0-15 = K k0..k0+15
// lanes 16-31 (M=lane-16): elems 0-15 = K k0+16..k0+31
__device__ __forceinline__ v16bf afrag(const __bf16* __restrict__ base, int ld,
                                       int k0, int lane) {
  const __bf16* p = base + (lane & 15) * ld + k0 + ((lane & 16) ? 16 : 0);
  v16bf a;
#pragma unroll
  for (int t = 0; t < 16; ++t) { a[t] = p[t]; }
  return a;
}

// ---------------------------------------------------------------------------
// Kernel 0: zero output + scratch header
// ---------------------------------------------------------------------------
__global__ void zero_kernel(float* __restrict__ out, int n_out,
                            int* __restrict__ hdr, int n_hdr) {
  int i = blockIdx.x * blockDim.x + threadIdx.x;
  int stride = gridDim.x * blockDim.x;
  for (int j = i; j < n_out; j += stride) out[j] = 0.0f;
  if (i < n_hdr) hdr[i] = 0;
}

// ---------------------------------------------------------------------------
// Kernel 0b: fp32 -> bf16 weight image (fast path only). n multiple of 4.
// ---------------------------------------------------------------------------
__global__ void cvt_kernel(const float* __restrict__ src,
                           __bf16* __restrict__ dst, int n4) {
  int i = blockIdx.x * blockDim.x + threadIdx.x;
  int stride = gridDim.x * blockDim.x;
  for (int j = i; j < n4; j += stride) {
    float4 f = ((const float4*)src)[j];
    v4bf o;
    o[0] = (__bf16)f.x; o[1] = (__bf16)f.y; o[2] = (__bf16)f.z; o[3] = (__bf16)f.w;
    ((v4bf*)dst)[j] = o;
  }
}

// ---------------------------------------------------------------------------
// Kernel 1: routing. One wave (32 lanes) per token.
// ---------------------------------------------------------------------------
__global__ __launch_bounds__(256) void route_kernel(
    const float* __restrict__ x,  const float* __restrict__ Wm,
    const float* __restrict__ bm, const float* __restrict__ Wr,
    const float* __restrict__ br,
    int* __restrict__ counts, int* __restrict__ cnt_g,
    float* __restrict__ sum_pm, float* __restrict__ sum_probs,
    int* __restrict__ list_tok, float* __restrict__ list_w) {
  int wave  = threadIdx.x >> 5;
  int lane  = threadIdx.x & 31;
  int token = blockIdx.x * (blockDim.x >> 5) + wave;
  if (token >= N_) return;

  const float* xr = x + (size_t)token * D_;
  float xl[16];
#pragma unroll
  for (int i = 0; i < 16; ++i) xl[i] = xr[lane + 32 * i];

  // meta logits (G=3)
  float ml[G_];
#pragma unroll
  for (int gi = 0; gi < G_; ++gi) {
    const float* wr = Wm + gi * D_;
    float p = 0.0f;
#pragma unroll
    for (int i = 0; i < 16; ++i) p += xl[i] * wr[lane + 32 * i];
#pragma unroll
    for (int off = 16; off > 0; off >>= 1) p += __shfl_xor(p, off, 32);
    ml[gi] = p + bm[gi];
  }
  int g = 0;
#pragma unroll
  for (int j = 1; j < G_; ++j) if (ml[j] > ml[g]) g = j;   // first-max, as argmax

  // selected group's router logits (E=4)
  float sl[E_];
#pragma unroll
  for (int e = 0; e < E_; ++e) {
    const float* wr = Wr + ((size_t)g * E_ + e) * D_;
    float p = 0.0f;
#pragma unroll
    for (int i = 0; i < 16; ++i) p += xl[i] * wr[lane + 32 * i];
#pragma unroll
    for (int off = 16; off > 0; off >>= 1) p += __shfl_xor(p, off, 32);
    sl[e] = p + br[g * E_ + e];
  }

  if (lane == 0) {
    // aux: mean softmax over meta logits
    float mx = fmaxf(ml[0], fmaxf(ml[1], ml[2]));
    float ex[G_], s = 0.0f;
#pragma unroll
    for (int j = 0; j < G_; ++j) { ex[j] = expf(ml[j] - mx); s += ex[j]; }
#pragma unroll
    for (int j = 0; j < G_; ++j) atomicAdd(&sum_pm[j], ex[j] / s);

    // aux: per-group mean expert probs
    float mx4 = fmaxf(fmaxf(sl[0], sl[1]), fmaxf(sl[2], sl[3]));
    float es[E_], ss = 0.0f;
#pragma unroll
    for (int e = 0; e < E_; ++e) { es[e] = expf(sl[e] - mx4); ss += es[e]; }
#pragma unroll
    for (int e = 0; e < E_; ++e) atomicAdd(&sum_probs[g * E_ + e], es[e] / ss);
    atomicAdd(&cnt_g[g], 1);

    // top-2 (lax.top_k tie-break: lower index first -> strict >)
    int i1 = 0;
#pragma unroll
    for (int e = 1; e < E_; ++e) if (sl[e] > sl[i1]) i1 = e;
    int i2 = (i1 == 0) ? 1 : 0;
#pragma unroll
    for (int e = 0; e < E_; ++e) if (e != i1 && sl[e] > sl[i2]) i2 = e;
    float e2 = expf(sl[i2] - sl[i1]);
    float w1 = 1.0f / (1.0f + e2);
    float w2 = e2 / (1.0f + e2);

    int p1 = g * E_ + i1;
    int pos1 = atomicAdd(&counts[p1], 1);
    list_tok[p1 * N_ + pos1] = token; list_w[p1 * N_ + pos1] = w1;
    int p2 = g * E_ + i2;
    int pos2 = atomicAdd(&counts[p2], 1);
    list_tok[p2 * N_ + pos2] = token; list_w[p2 * N_ + pos2] = w2;
  }
}

// ---------------------------------------------------------------------------
// Kernel 2: fold aux statistics into the scalar output slot.
// ---------------------------------------------------------------------------
__global__ void aux_kernel(const int* __restrict__ cnt_g,
                           const float* __restrict__ sum_pm,
                           const float* __restrict__ sum_probs,
                           float* __restrict__ out_aux) {
  if (threadIdx.x != 0 || blockIdx.x != 0) return;
  float s2 = 0.0f;
  for (int j = 0; j < G_; ++j) { float pm = sum_pm[j] / (float)N_; s2 += pm * pm; }
  float aux = ALPHA_ * (float)G_ * s2;
  for (int g = 0; g < G_; ++g) {
    int c = cnt_g[g];
    if (c > 0) {
      float denom = (float)(c < 1 ? 1 : c);
      float t = 0.0f;
      for (int e = 0; e < E_; ++e) {
        float m = sum_probs[g * E_ + e] / denom;
        t += m * m;
      }
      aux += ALPHA_ * (float)E_ * t;
    }
  }
  *out_aux = aux;
}

// ---------------------------------------------------------------------------
// Kernel 3: fused expert MLP on a 16-token tile, bf16 WMMA.
//   h  = relu(x@W1^T + b1)         [16,1024]
//   hr = relu(h@W2^T + b2 + x@Wp^T + bp)
//   y  = hr@W3^T + b3              [16,512]
//   out[token] += w * y            (atomic; each token is in exactly 2 lists)
// Templated on weight element type: __bf16 (pre-converted) or float (fallback).
// ---------------------------------------------------------------------------
template <typename WT>
__global__ __launch_bounds__(256) void expert_kernel(
    const float* __restrict__ x,
    const WT* __restrict__ W1, const float* __restrict__ b1,
    const WT* __restrict__ W2, const float* __restrict__ b2,
    const WT* __restrict__ Wp, const float* __restrict__ bp,
    const WT* __restrict__ W3, const float* __restrict__ b3,
    const int* __restrict__ counts, const int* __restrict__ list_tok,
    const float* __restrict__ list_w, float* __restrict__ out) {
  constexpr int XS_LD    = D_ + 8;                 // 520 (keeps 16B alignment)
  constexpr int H_LD     = H_ + 8;                 // 1032
  constexpr int XS_BYTES = 16 * XS_LD * 2;         // 16640
  constexpr int HS_BYTES = 16 * H_LD * 2;          // 33024
  __shared__ __align__(16) char smem[XS_BYTES + HS_BYTES];   // ~49.7 KB
  __shared__ int   tok_s[16];
  __shared__ float w_s[16];

  const int pair = blockIdx.y;                     // (group, expert) flat id
  const int cnt  = counts[pair];
  const int base = blockIdx.x * 16;
  if (base >= cnt) return;                         // uniform branch

  const int tid  = threadIdx.x;
  const int lane = tid & 31;
  const int wave = tid >> 5;

  if (tid < 16) {
    int idx = base + tid;
    if (idx < cnt) {
      tok_s[tid] = list_tok[pair * N_ + idx];
      w_s[tid]   = list_w[pair * N_ + idx];
    } else {
      tok_s[tid] = -1;
      w_s[tid]   = 0.0f;
    }
  }
  __syncthreads();

  __bf16* xs = (__bf16*)smem;
  __bf16* hs = (__bf16*)(smem + XS_BYTES);

  // gather x tile -> LDS (bf16, row-major)
  for (int i = tid; i < 16 * D_; i += 256) {
    int r = i >> 9, c = i & (D_ - 1);
    int t = tok_s[r];
    float v = (t >= 0) ? x[(size_t)t * D_ + c] : 0.0f;
    xs[r * XS_LD + c] = (__bf16)v;
  }
  __syncthreads();

  const WT* W1p = W1 + (size_t)pair * H_ * D_;
  const float* b1p = b1 + pair * H_;
  const WT* W2p = W2 + (size_t)pair * H_ * H_;
  const float* b2p = b2 + pair * H_;
  const WT* Wpp = Wp + (size_t)pair * H_ * D_;
  const float* bpp = bp + pair * H_;
  const WT* W3p = W3 + (size_t)pair * O_ * H_;
  const float* b3p = b3 + pair * O_;

  const v8f vzero = {};

  // ---- GEMM1: h = relu(x @ W1^T + b1), each wave owns 128 columns ----
  {
    v8f acc[8];
#pragma unroll
    for (int nt = 0; nt < 8; ++nt) acc[nt] = vzero;
    for (int k0 = 0; k0 < D_; k0 += 32) {
      v16bf a = afrag(xs, XS_LD, k0, lane);
#pragma unroll
      for (int nt = 0; nt < 8; ++nt) {
        int n = wave * 128 + nt * 16 + (lane & 15);
        v16bf b = bfrag(W1p + (size_t)n * D_ + k0 + ((lane & 16) ? 16 : 0));
        acc[nt] = wmma_bf16(a, b, acc[nt]);
      }
    }
#pragma unroll
    for (int nt = 0; nt < 8; ++nt) {
      int n = wave * 128 + nt * 16 + (lane & 15);
      float bias = b1p[n];
#pragma unroll
      for (int r = 0; r < 8; ++r) {
        int m = r + ((lane & 16) ? 8 : 0);
        hs[m * H_LD + n] = (__bf16)fmaxf(acc[nt][r] + bias, 0.0f);
      }
    }
  }
  __syncthreads();

  // ---- GEMM2 + residual projection: hr = relu(h@W2^T + b2 + x@Wp^T + bp) ----
  v8f acc2[8];
#pragma unroll
  for (int nt = 0; nt < 8; ++nt) acc2[nt] = vzero;
  for (int k0 = 0; k0 < H_; k0 += 32) {
    v16bf a = afrag(hs, H_LD, k0, lane);
#pragma unroll
    for (int nt = 0; nt < 8; ++nt) {
      int n = wave * 128 + nt * 16 + (lane & 15);
      v16bf b = bfrag(W2p + (size_t)n * H_ + k0 + ((lane & 16) ? 16 : 0));
      acc2[nt] = wmma_bf16(a, b, acc2[nt]);
    }
  }
  for (int k0 = 0; k0 < D_; k0 += 32) {
    v16bf a = afrag(xs, XS_LD, k0, lane);
#pragma unroll
    for (int nt = 0; nt < 8; ++nt) {
      int n = wave * 128 + nt * 16 + (lane & 15);
      v16bf b = bfrag(Wpp + (size_t)n * D_ + k0 + ((lane & 16) ? 16 : 0));
      acc2[nt] = wmma_bf16(a, b, acc2[nt]);
    }
  }
  __syncthreads();                 // all reads of xs/hs complete
  __bf16* hr = (__bf16*)smem;      // hr overlays the xs/hs region (33 KB)
#pragma unroll
  for (int nt = 0; nt < 8; ++nt) {
    int n = wave * 128 + nt * 16 + (lane & 15);
    float bias = b2p[n] + bpp[n];
#pragma unroll
    for (int r = 0; r < 8; ++r) {
      int m = r + ((lane & 16) ? 8 : 0);
      hr[m * H_LD + n] = (__bf16)fmaxf(acc2[nt][r] + bias, 0.0f);
    }
  }
  __syncthreads();

  // ---- GEMM3: y = hr @ W3^T + b3, each wave owns 64 columns of O=512 ----
  {
    v8f acc[4];
#pragma unroll
    for (int nt = 0; nt < 4; ++nt) acc[nt] = vzero;
    for (int k0 = 0; k0 < H_; k0 += 32) {
      v16bf a = afrag(hr, H_LD, k0, lane);
#pragma unroll
      for (int nt = 0; nt < 4; ++nt) {
        int n = wave * 64 + nt * 16 + (lane & 15);
        v16bf b = bfrag(W3p + (size_t)n * H_ + k0 + ((lane & 16) ? 16 : 0));
        acc[nt] = wmma_bf16(a, b, acc[nt]);
      }
    }
#pragma unroll
    for (int nt = 0; nt < 4; ++nt) {
      int n = wave * 64 + nt * 16 + (lane & 15);
      float bias = b3p[n];
#pragma unroll
      for (int r = 0; r < 8; ++r) {
        int m = r + ((lane & 16) ? 8 : 0);
        int t = tok_s[m];
        if (t >= 0)
          atomicAdd(&out[(size_t)t * O_ + n], w_s[m] * (acc[nt][r] + bias));
      }
    }
  }
}

// ---------------------------------------------------------------------------
// Host launcher
// ---------------------------------------------------------------------------
extern "C" void kernel_launch(void* const* d_in, const int* in_sizes, int n_in,
                              void* d_out, int out_size, void* d_ws, size_t ws_size,
                              hipStream_t stream) {
  const float* x  = (const float*)d_in[0];
  const float* Wm = (const float*)d_in[1];
  const float* bm = (const float*)d_in[2];
  const float* Wr = (const float*)d_in[3];
  const float* br = (const float*)d_in[4];
  const float* W1 = (const float*)d_in[5];
  const float* b1 = (const float*)d_in[6];
  const float* W2 = (const float*)d_in[7];
  const float* b2 = (const float*)d_in[8];
  const float* Wp = (const float*)d_in[9];
  const float* bp = (const float*)d_in[10];
  const float* W3 = (const float*)d_in[11];
  const float* b3 = (const float*)d_in[12];
  float* out = (float*)d_out;

  // scratch layout (32-word header + token/weight lists + optional bf16 weights)
  int*   ws_i      = (int*)d_ws;
  int*   counts    = ws_i;            // [12]
  int*   cnt_g     = ws_i + 12;       // [3] (padded to 16)
  float* sum_pm    = (float*)(ws_i + 16);  // [3] (padded to 4)
  float* sum_probs = (float*)(ws_i + 20);  // [12]
  int*   list_tok  = ws_i + 32;                      // [12*N]
  float* list_w    = (float*)(ws_i + 32 + P_ * N_);  // [12*N]

  const size_t header_bytes = (size_t)(32 + 2 * P_ * N_) * 4;
  const size_t wt_off       = (header_bytes + 255) & ~(size_t)255;
  const size_t n_w1 = (size_t)P_ * H_ * D_;   // 6,291,456
  const size_t n_w2 = (size_t)P_ * H_ * H_;   // 12,582,912
  const size_t n_wp = (size_t)P_ * H_ * D_;
  const size_t n_w3 = (size_t)P_ * O_ * H_;
  const size_t n_wt = n_w1 + n_w2 + n_wp + n_w3;
  const bool use_bf16 = ws_size >= wt_off + n_wt * sizeof(__bf16);

  const int n_out = N_ * O_;

  zero_kernel<<<4096, 256, 0, stream>>>(out, n_out, ws_i, 32);

  route_kernel<<<N_ / 8, 256, 0, stream>>>(x, Wm, bm, Wr, br,
                                           counts, cnt_g, sum_pm, sum_probs,
                                           list_tok, list_w);

  aux_kernel<<<1, 1, 0, stream>>>(cnt_g, sum_pm, sum_probs, out + n_out);

  dim3 grid(N_ / 16, P_);
  if (use_bf16) {
    __bf16* W1b = (__bf16*)((char*)d_ws + wt_off);
    __bf16* W2b = W1b + n_w1;
    __bf16* Wpb = W2b + n_w2;
    __bf16* W3b = Wpb + n_wp;
    cvt_kernel<<<2048, 256, 0, stream>>>(W1, W1b, (int)(n_w1 / 4));
    cvt_kernel<<<2048, 256, 0, stream>>>(W2, W2b, (int)(n_w2 / 4));
    cvt_kernel<<<2048, 256, 0, stream>>>(Wp, Wpb, (int)(n_wp / 4));
    cvt_kernel<<<2048, 256, 0, stream>>>(W3, W3b, (int)(n_w3 / 4));
    expert_kernel<__bf16><<<grid, 256, 0, stream>>>(
        x, W1b, b1, W2b, b2, Wpb, bp, W3b, b3, counts, list_tok, list_w, out);
  } else {
    expert_kernel<float><<<grid, 256, 0, stream>>>(
        x, W1, b1, W2, b2, Wp, bp, W3, b3, counts, list_tok, list_w, out);
  }
}
